// LSTMCell_33526514712649
// MI455X (gfx1250) — matrix operand order repaired and checked by reading, not compiled, not run on hardware
//
#include <hip/hip_runtime.h>
#include <stddef.h>
#include <stdint.h>

// ---------------------------------------------------------------------------
// LSTM cell fused kernel for gfx1250 (MI455X).
//   gates = [x|h] @ [W;U]  (bf16 WMMA, f32 accum), fused gate math -> h_t
// Workspace layout (needs 32 MB):
//   [0    .. 16MB) : A  bf16 [4096][2048]    = [x_t | h_prev]
//   [16MB .. 32MB) : Bt bf16 [4][1024][2048] = per-gate [W;U] transposed
// GEMM kernel: 64(M) x 64(N) tile per 256-thread WG, 4 gates fused,
// double-buffered LDS filled with global_load_async_to_lds_b128 (saddr form,
// ASYNCcnt), pipeline unrolled 2 stages so buffer parity is compile-time.
// ---------------------------------------------------------------------------

typedef __attribute__((ext_vector_type(16))) __bf16 v16bf;
typedef __attribute__((ext_vector_type(8)))  float  v8f;
typedef unsigned short u16;
typedef unsigned int   u32;

union Frag {
    v16bf v;
    uint4 q[2];
};

__device__ __forceinline__ u16 f2bf(float f) {
    u32 u = __float_as_uint(f);
    u32 r = (u + 0x7FFFu + ((u >> 16) & 1u)) >> 16;   // round-to-nearest-even
    return (u16)r;
}
__device__ __forceinline__ u32 pack2bf(float lo, float hi) {
    return (u32)f2bf(lo) | ((u32)f2bf(hi) << 16);
}
__device__ __forceinline__ float fast_sigmoid(float x) {
    return 1.0f / (1.0f + __expf(-x));
}
__device__ __forceinline__ float fast_tanh(float x) {
    return 1.0f - 2.0f / (__expf(2.0f * x) + 1.0f);   // saturates correctly
}

// -------- conversion kernel: A = [x | h] -> bf16, 8 elems/thread -----------
__global__ __launch_bounds__(256)
void lstm_convA(const float* __restrict__ x, const float* __restrict__ h,
                u16* __restrict__ A) {
    int idx = (blockIdx.x * 256 + threadIdx.x) * 8;   // element index, k-contig
    int m = idx >> 11;
    int k = idx & 2047;                               // 8 | k, never straddles 1024
    const float* src = (k < 1024) ? (x + m * 1024 + k) : (h + m * 1024 + k - 1024);
    float4 a = ((const float4*)src)[0];
    float4 b = ((const float4*)src)[1];
    uint4 o;
    o.x = pack2bf(a.x, a.y);
    o.y = pack2bf(a.z, a.w);
    o.z = pack2bf(b.x, b.y);
    o.w = pack2bf(b.z, b.w);
    *(uint4*)(A + idx) = o;
}

// -------- conversion kernel: Bt[g][n][k] = ([W_g;U_g])^T, LDS transpose ----
__global__ __launch_bounds__(256)
void lstm_convB(const float* __restrict__ W_f, const float* __restrict__ W_i,
                const float* __restrict__ W_c, const float* __restrict__ W_o,
                const float* __restrict__ U_f, const float* __restrict__ U_i,
                const float* __restrict__ U_c, const float* __restrict__ U_o,
                u16* __restrict__ Bt) {
    __shared__ u16 T[32 * 33];                    // +1 pad: conflict-free transpose
    int g   = blockIdx.x >> 11;                   // 4 gates
    int rem = blockIdx.x & 2047;
    int kt  = rem >> 5;                           // 64 k-tiles of 32 (K = 2048)
    int nt  = rem & 31;                           // 32 n-tiles of 32 (N = 1024)
    const float* Wg = (g == 0) ? W_f : (g == 1) ? W_i : (g == 2) ? W_c : W_o;
    const float* Ug = (g == 0) ? U_f : (g == 1) ? U_i : (g == 2) ? U_c : U_o;

    #pragma unroll
    for (int j = 0; j < 4; ++j) {                 // load 32x32 f32 tile, n contiguous
        int e  = threadIdx.x + j * 256;
        int kk = e >> 5, nn = e & 31;
        int k  = kt * 32 + kk, n = nt * 32 + nn;
        float v = (k < 1024) ? Wg[k * 1024 + n] : Ug[(k - 1024) * 1024 + n];
        T[kk * 33 + nn] = f2bf(v);
    }
    __syncthreads();
    #pragma unroll
    for (int j = 0; j < 4; ++j) {                 // store transposed, k contiguous
        int e  = threadIdx.x + j * 256;
        int nn = e >> 5, kk = e & 31;
        Bt[((g * 1024) + nt * 32 + nn) * 2048 + kt * 32 + kk] = T[kk * 33 + nn];
    }
}

// -------- fused GEMM + gate kernel -----------------------------------------
// 256 threads = 8 waves: wm = wave&1 (32-row M half), wn = wave>>1 (16-col N).
// Each wave: 2 M-subtiles x 4 gates = 8 accumulators; 12 ds_load -> 8 WMMA /
// K-step. LDS double buffer filled via global_load_async_to_lds_b128 (saddr).
#define KT    32
#define LDSU  (2048 + 4 * 64 * 32)     // u16 per buffer: As 64x32 + Bs 4x64x32

__global__ __launch_bounds__(256)
void lstm_gemm(const u16* __restrict__ A,        // [4096][2048] bf16
               const u16* __restrict__ Bt,       // [4][1024][2048] bf16
               const float* __restrict__ c_prev, // [4096][1024]
               const float* __restrict__ b_f, const float* __restrict__ b_i,
               const float* __restrict__ b_c, const float* __restrict__ b_o,
               float* __restrict__ h_out)        // [4096][1024]
{
    __shared__ u16 lds[2][LDSU];                 // 2 x 20 KB

    const int tid  = threadIdx.x;
    const int lane = tid & 31;
    const int wave = tid >> 5;
    const int wm   = wave & 1;                   // M half (32 rows)
    const int wn   = wave >> 1;                  // N quarter (16 cols)
    const int M0   = blockIdx.x * 64;            // gridDim.x = 64
    const int N0   = blockIdx.y * 64;            // gridDim.y = 16

    // ---- per-thread async staging descriptors (5 x 16B segments/stage) ----
    // j=0   : A tile  64x32   (256 segs, 1/thread)  off goffA from A base
    // j=1..4: B tiles 4x64x32 (1024 segs, 4/thread) off goffB[] from Bt base
    // Offsets are 32-bit bytes (GVS addressing: SGPR64 base + VGPR32 offset),
    // advanced by +64 bytes per issued stage.
    u32 goffA;
    u32 goffB[4];
    u32 l0[5], l1[5];                            // LDS byte addresses per buffer
    {
        const u32 base0 = (u32)(uintptr_t)(&lds[0][0]);
        int row = tid >> 2, sg = tid & 3;
        goffA = (u32)(((M0 + row) * 2048 + sg * 8) * 2);
        l0[0] = base0 + (u32)((row * 32 + sg * 8) * 2);
        #pragma unroll
        for (int j = 1; j < 5; ++j) {
            int s = (j - 1) * 256 + tid;
            int g = s >> 8, r = s & 255;
            int n = r >> 2, sb = r & 3;
            goffB[j - 1] = (u32)(((g * 1024 + N0 + n) * 2048 + sb * 8) * 2);
            l0[j] = base0 + (u32)((2048 + g * 2048 + n * 32 + sb * 8) * 2);
        }
        #pragma unroll
        for (int j = 0; j < 5; ++j) l1[j] = l0[j] + (u32)(LDSU * 2);
    }

    auto issue_stage = [&](const u32* l) {
        asm volatile("global_load_async_to_lds_b128 %0, %1, %2"
                     :: "v"(l[0]), "v"(goffA), "s"(A) : "memory");
        goffA += 2 * KT;
        #pragma unroll
        for (int j = 0; j < 4; ++j) {
            asm volatile("global_load_async_to_lds_b128 %0, %1, %2"
                         :: "v"(l[j + 1]), "v"(goffB[j]), "s"(Bt) : "memory");
            goffB[j] += 2 * KT;
        }
    };

    // ---- fragment addressing (ISA 16-bit WMMA layouts) ----
    const int arow = wm * 32 + (lane & 15);      // + msub*16
    const int akb  = (lane >> 4) * 8;
    const int bn   = wn * 16 + (lane & 15);
    const int bkb  = (lane >> 4) * 8;

    v8f acc[4][2] = {};                          // [gate][msub]

    auto compute_stage = [&](const u16* Ls) {
        const u16* As = Ls;
        const u16* Bs = Ls + 2048;
        Frag fa0, fa1;
        fa0.q[0] = *(const uint4*)(As + arow * 32 + akb);
        fa0.q[1] = *(const uint4*)(As + arow * 32 + 16 + akb);
        fa1.q[0] = *(const uint4*)(As + (arow + 16) * 32 + akb);
        fa1.q[1] = *(const uint4*)(As + (arow + 16) * 32 + 16 + akb);
        #pragma unroll
        for (int g = 0; g < 4; ++g) {
            Frag fb;
            fb.q[0] = *(const uint4*)(Bs + g * 2048 + bn * 32 + bkb);
            fb.q[1] = *(const uint4*)(Bs + g * 2048 + bn * 32 + bkb + 16);
            acc[g][0] = __builtin_amdgcn_wmma_f32_16x16x32_bf16(
                false, fa0.v, false, fb.v, (short)0, acc[g][0], false, false);
            acc[g][1] = __builtin_amdgcn_wmma_f32_16x16x32_bf16(
                false, fa1.v, false, fb.v, (short)0, acc[g][1], false, false);
        }
    };

    // ---- software pipeline: 2 stages per body, buffer parity is static ----
    issue_stage(l0);                             // stage k=0 -> buf0
    for (int k0 = 0; k0 < 2048; k0 += 2 * KT) {
        issue_stage(l1);                         // stage k0+32 -> buf1
        asm volatile("s_wait_asynccnt 0x5" ::: "memory");   // buf0 resident
        __syncthreads();
        compute_stage(lds[0]);
        __syncthreads();                         // all reads of buf0 done

        if (k0 + 2 * KT < 2048) {
            issue_stage(l0);                     // stage k0+64 -> buf0
            asm volatile("s_wait_asynccnt 0x5" ::: "memory"); // buf1 resident
        } else {
            asm volatile("s_wait_asynccnt 0x0" ::: "memory");
        }
        __syncthreads();
        compute_stage(lds[1]);
        __syncthreads();                         // all reads of buf1 done
    }

    // ---- epilogue: bias + gates + state update, write h_t ----
    // C/D layout: lane l, VGPR r -> m = (l>>4)*8 + r, n = l&15
    const int N = N0 + wn * 16 + (lane & 15);
    const float bF = b_f[N], bI = b_i[N], bC = b_c[N], bO = b_o[N];

    #pragma unroll
    for (int msub = 0; msub < 2; ++msub) {
        const int mbase = M0 + wm * 32 + msub * 16 + (lane >> 4) * 8;
        #pragma unroll
        for (int r = 0; r < 8; ++r) {
            int M = mbase + r;
            float f  = fast_sigmoid(acc[0][msub][r] + bF);
            float i  = fast_sigmoid(acc[1][msub][r] + bI);
            float ct = fast_tanh   (acc[2][msub][r] + bC);
            float o  = fast_sigmoid(acc[3][msub][r] + bO);
            float c  = f * c_prev[(size_t)M * 1024 + N] + i * ct;
            h_out[(size_t)M * 1024 + N] = o * fast_tanh(c);
        }
    }
}

// ---------------------------------------------------------------------------
extern "C" void kernel_launch(void* const* d_in, const int* in_sizes, int n_in,
                              void* d_out, int out_size, void* d_ws, size_t ws_size,
                              hipStream_t stream) {
    (void)in_sizes; (void)n_in; (void)out_size; (void)ws_size;

    const float* x_t    = (const float*)d_in[0];
    const float* h_prev = (const float*)d_in[1];
    const float* c_prev = (const float*)d_in[2];
    const float* W_f = (const float*)d_in[3];
    const float* W_i = (const float*)d_in[4];
    const float* W_c = (const float*)d_in[5];
    const float* W_o = (const float*)d_in[6];
    const float* U_f = (const float*)d_in[7];
    const float* U_i = (const float*)d_in[8];
    const float* U_c = (const float*)d_in[9];
    const float* U_o = (const float*)d_in[10];
    const float* b_f = (const float*)d_in[11];
    const float* b_i = (const float*)d_in[12];
    const float* b_c = (const float*)d_in[13];
    const float* b_o = (const float*)d_in[14];

    u16* Abf  = (u16*)d_ws;                          // 4096*2048 bf16 = 16 MB
    u16* Btbf = Abf + (size_t)4096 * 2048;           // 4*1024*2048 bf16 = 16 MB

    lstm_convA<<<(4096 * 2048) / (256 * 8), 256, 0, stream>>>(x_t, h_prev, Abf);
    lstm_convB<<<4 * 64 * 32, 256, 0, stream>>>(W_f, W_i, W_c, W_o,
                                                U_f, U_i, U_c, U_o, Btbf);

    dim3 grid(64, 16);   // 64 M-tiles x 16 N-tiles of 64x64
    lstm_gemm<<<grid, 256, 0, stream>>>(Abf, Btbf, c_prev,
                                        b_f, b_i, b_c, b_o, (float*)d_out);
}
